// WindowMultiHeadAttention_89687507075583
// MI455X (gfx1250) — compile-verified, matrix operand order checked by reading, not run on hardware
//
#include <hip/hip_runtime.h>
#include <hip/hip_bf16.h>

// ---------------------------------------------------------------------------
// MI455X (gfx1250) fused Swin-v2 window attention.
// All GEMMs via v_wmma_f32_16x16x32_f16 (f16 in, f32 accum), wave32.
// ---------------------------------------------------------------------------

typedef __attribute__((ext_vector_type(16))) _Float16 v16h;
typedef __attribute__((ext_vector_type(8)))  float    v8f;

#define BW   256
#define LL   256
#define CC   512
#define HH   16
#define DD   32
#define NQKV 1536
#define LOG_MAX 4.605170185988092f

__device__ __forceinline__ v8f wmma16(v16h a, v16h b, v8f c) {
    return __builtin_amdgcn_wmma_f32_16x16x32_f16(
        false, a, false, b, (short)0, c, false, false);
}

// Build a v16h fragment from two 16-byte chunks (global or LDS).
// CDNA5 16-bit A/B layout: lane L holds 16 halfs; K offsets are
// { hl*8 .. hl*8+7 , 16+hl*8 .. 16+hl*8+7 } within a 32-wide K slice,
// where hl = (lane>=16). Both chunks are 16B-aligned contiguous runs.
__device__ __forceinline__ v16h frag_from(const _Float16* p0, const _Float16* p1) {
    v16h r;
    ((float4*)&r)[0] = *(const float4*)p0;
    ((float4*)&r)[1] = *(const float4*)p1;
    return r;
}

// ---------------------------------------------------------------------------
// 1) f32 -> f16 convert (4 elements / thread, sizes are multiples of 4)
// ---------------------------------------------------------------------------
typedef __attribute__((ext_vector_type(4))) _Float16 v4h;

__global__ __launch_bounds__(256) void convert_f16_kernel(
    const float* __restrict__ in, _Float16* __restrict__ out, size_t n4) {
    size_t i = (size_t)blockIdx.x * 256 + threadIdx.x;
    if (i >= n4) return;
    float4 v = ((const float4*)in)[i];
    v4h h; h.x = (_Float16)v.x; h.y = (_Float16)v.y;
    h.z = (_Float16)v.z; h.w = (_Float16)v.w;
    ((v4h*)out)[i] = h;
}

// ---------------------------------------------------------------------------
// 2) Relative-position-bias MLP -> table[16][961]
//    (bias depends only on (dy,dx) in [-15,15]^2 -> 961 unique entries)
// ---------------------------------------------------------------------------
__global__ __launch_bounds__(512) void bias_table_kernel(
    const float* __restrict__ fc1_w, const float* __restrict__ fc1_b,
    const float* __restrict__ fc2_w, const float* __restrict__ fc2_b,
    float* __restrict__ table) {
    __shared__ float hid[512];
    int e  = blockIdx.x;            // 0..960
    int dy = e / 31 - 15, dx = e % 31 - 15;
    float r0 = (dy > 0 ? 1.f : (dy < 0 ? -1.f : 0.f)) * log1pf(fabsf((float)dy));
    float r1 = (dx > 0 ? 1.f : (dx < 0 ? -1.f : 0.f)) * log1pf(fabsf((float)dx));
    int t = threadIdx.x;
    hid[t] = fmaxf(fc1_w[t * 2 + 0] * r0 + fc1_w[t * 2 + 1] * r1 + fc1_b[t], 0.f);
    __syncthreads();
    int w = t >> 5, lane = t & 31;  // 16 waves == 16 heads
    float s = 0.f;
    #pragma unroll
    for (int k = lane; k < 512; k += 32) s += hid[k] * fc2_w[w * 512 + k];
    #pragma unroll
    for (int m = 16; m > 0; m >>= 1) s += __shfl_xor(s, m, 32);
    if (lane == 0) table[w * 961 + e] = s + fc2_b[w];
}

// ---------------------------------------------------------------------------
// 3) Tiled f16 GEMM:  C(MxN) = A(MxK) * Bt(NxK)^T + bias
//    block tile 128x128, 8 waves (4x2) of 32x64, K-step 32.
//    mode 0: f32 row-major output.  mode 1: scatter qkv into packed f16
//    q/k (head-major rows, d contiguous) and vT (d-major rows, L contiguous).
// ---------------------------------------------------------------------------
__global__ __launch_bounds__(256) void gemm_f16_kernel(
    const _Float16* __restrict__ A,  const _Float16* __restrict__ Bt,
    const float* __restrict__ bias,  float* __restrict__ Cout,
    _Float16* __restrict__ qh, _Float16* __restrict__ kh,
    _Float16* __restrict__ vT, int M, int N, int K, int mode) {
    __shared__ _Float16 As[128 * 32];
    __shared__ _Float16 Bs[128 * 32];
    const int tid = threadIdx.x;
    const int wid = tid >> 5, lane = tid & 31;
    const int hl = lane >> 4, l15 = lane & 15;
    const int bm = blockIdx.x * 128, bn = blockIdx.y * 128;
    const int wm = (wid & 3) * 32, wn = (wid >> 2) * 64;

    v8f acc[2][4];
    #pragma unroll
    for (int i = 0; i < 2; ++i)
        #pragma unroll
        for (int j = 0; j < 4; ++j) acc[i][j] = (v8f){};

    for (int k0 = 0; k0 < K; k0 += 32) {
        // stage 128x32 A and 128x32 Bt slices (16B chunks, 2 per thread each)
        #pragma unroll
        for (int c = tid; c < 512; c += 256) {
            int row = c >> 2, seg = (c & 3) * 8;
            *(float4*)&As[row * 32 + seg] =
                *(const float4*)&A[(size_t)(bm + row) * K + k0 + seg];
            *(float4*)&Bs[row * 32 + seg] =
                *(const float4*)&Bt[(size_t)(bn + row) * K + k0 + seg];
        }
        __syncthreads();
        v16h af[2], bf[4];
        #pragma unroll
        for (int i = 0; i < 2; ++i) {
            const _Float16* r = &As[(wm + i * 16 + l15) * 32];
            af[i] = frag_from(r + hl * 8, r + 16 + hl * 8);
        }
        #pragma unroll
        for (int j = 0; j < 4; ++j) {
            const _Float16* r = &Bs[(wn + j * 16 + l15) * 32];
            bf[j] = frag_from(r + hl * 8, r + 16 + hl * 8);
        }
        #pragma unroll
        for (int i = 0; i < 2; ++i)
            #pragma unroll
            for (int j = 0; j < 4; ++j) acc[i][j] = wmma16(af[i], bf[j], acc[i][j]);
        __syncthreads();
    }

    #pragma unroll
    for (int i = 0; i < 2; ++i)
        #pragma unroll
        for (int j = 0; j < 4; ++j)
            #pragma unroll
            for (int r = 0; r < 8; ++r) {
                int grow = bm + wm + i * 16 + r + hl * 8;
                int gcol = bn + wn + j * 16 + l15;
                float v = acc[i][j][r] + bias[gcol];
                if (mode == 0) {
                    Cout[(size_t)grow * N + gcol] = v;
                } else {
                    int three = gcol >> 9, h = (gcol >> 5) & 15, dd = gcol & 31;
                    int b = grow >> 8, l = grow & 255;
                    _Float16 hv = (_Float16)v;
                    if (three == 2)
                        vT[((size_t)(b * HH + h) * DD + dd) * LL + l] = hv;
                    else {
                        size_t o = ((size_t)(b * HH + h) * LL + l) * DD + dd;
                        (three == 0 ? qh : kh)[o] = hv;
                    }
                }
            }
}

// ---------------------------------------------------------------------------
// 4) L2-normalize 32-wide rows in place; optionally fold per-head logit scale
//    (one wave per row, 8 rows per block)
// ---------------------------------------------------------------------------
__global__ __launch_bounds__(256) void normalize_kernel(
    _Float16* __restrict__ t, const float* __restrict__ logit_scale, int useScale) {
    size_t row = (size_t)blockIdx.x * 8 + (threadIdx.x >> 5);
    int lane = threadIdx.x & 31;
    _Float16* p = t + row * DD;
    float v = (float)p[lane];
    float s = v * v;
    #pragma unroll
    for (int m = 16; m > 0; m >>= 1) s += __shfl_xor(s, m, 32);
    float inv = 1.f / fmaxf(sqrtf(s), 1e-12f);
    if (useScale) {
        int h = (int)((row >> 8) & 15);           // row = (b*16+h)*256 + l
        inv *= __expf(fminf(logit_scale[h], LOG_MAX));
    }
    p[lane] = (_Float16)(v * inv);
}

// ---------------------------------------------------------------------------
// 5) Fused attention: one block per (b,h). K(256x32) and vT(32x256) in LDS,
//    exact softmax in registers, P streamed through 1KB/wave LDS staging.
// ---------------------------------------------------------------------------
__global__ __launch_bounds__(256) void attn_kernel(
    const _Float16* __restrict__ qh, const _Float16* __restrict__ kh,
    const _Float16* __restrict__ vT, const float* __restrict__ btab,
    _Float16* __restrict__ y) {
    __shared__ _Float16 Ks[LL * DD];        // 16 KB: key rows x d
    __shared__ _Float16 Vs[DD * LL];        // 16 KB: vT rows (dd) x L
    __shared__ _Float16 Pw[8][16 * 32];     //  8 KB: per-wave P staging
    __shared__ float    bh[961];            //  ~4 KB: per-head bias table
    const int bhid = blockIdx.x;            // b*16 + h
    const int h = bhid & 15, b = bhid >> 4;
    const int tid = threadIdx.x, wid = tid >> 5, lane = tid & 31;
    const int hl = lane >> 4, l15 = lane & 15;

    const _Float16* kg = kh + (size_t)bhid * LL * DD;
    const _Float16* vg = vT + (size_t)bhid * DD * LL;
    #pragma unroll
    for (int c = tid; c < 1024; c += 256) {
        *(float4*)&Ks[c * 8] = *(const float4*)&kg[c * 8];
        *(float4*)&Vs[c * 8] = *(const float4*)&vg[c * 8];
    }
    for (int c = tid; c < 961; c += 256) bh[c] = btab[h * 961 + c];
    __syncthreads();

    const _Float16* qg = qh + (size_t)bhid * LL * DD;
    for (int rg = 0; rg < 2; ++rg) {
        const int rbase = wid * 32 + rg * 16;   // this wave's 16 query rows
        // q fragment straight from global (16x32, d == one K-step)
        const _Float16* qr = qg + (size_t)(rbase + l15) * DD;
        v16h aq = frag_from(qr + hl * 8, qr + 16 + hl * 8);

        // S = q k^T : 16 WMMAs cover 16x256 scores
        v8f s[16];
        #pragma unroll
        for (int nt = 0; nt < 16; ++nt) {
            const _Float16* kr = &Ks[(nt * 16 + l15) * DD];
            v16h bk = frag_from(kr + hl * 8, kr + 16 + hl * 8);
            s[nt] = wmma16(aq, bk, (v8f){});
        }
        // + relative position bias (gathered from 961-entry table)
        #pragma unroll
        for (int nt = 0; nt < 16; ++nt) {
            int col = nt * 16 + l15, cy = col >> 4, cx = col & 15;
            #pragma unroll
            for (int r = 0; r < 8; ++r) {
                int row = rbase + r + hl * 8;
                s[nt][r] += bh[((row >> 4) - cy + 15) * 31 + ((row & 15) - cx + 15)];
            }
        }
        // exact softmax: reduce across 16 tiles then across the 16-lane
        // column group (xor masks 1..8 keep us inside each half-wave)
        float mx[8], sm[8], inv[8];
        #pragma unroll
        for (int r = 0; r < 8; ++r) mx[r] = -1e30f;
        #pragma unroll
        for (int nt = 0; nt < 16; ++nt)
            #pragma unroll
            for (int r = 0; r < 8; ++r) mx[r] = fmaxf(mx[r], s[nt][r]);
        #pragma unroll
        for (int r = 0; r < 8; ++r)
            #pragma unroll
            for (int m = 8; m > 0; m >>= 1) mx[r] = fmaxf(mx[r], __shfl_xor(mx[r], m, 32));
        #pragma unroll
        for (int r = 0; r < 8; ++r) sm[r] = 0.f;
        #pragma unroll
        for (int nt = 0; nt < 16; ++nt)
            #pragma unroll
            for (int r = 0; r < 8; ++r) {
                float e = __expf(s[nt][r] - mx[r]);
                s[nt][r] = e; sm[r] += e;
            }
        #pragma unroll
        for (int r = 0; r < 8; ++r) {
            #pragma unroll
            for (int m = 8; m > 0; m >>= 1) sm[r] += __shfl_xor(sm[r], m, 32);
            inv[r] = 1.f / sm[r];
        }

        // O = P V : stream P in 16x32 chunks via same-wave LDS (DS is
        // in-order per wave, no barrier needed), 2 WMMAs per chunk.
        v8f o[2] = {(v8f){}, (v8f){}};
        _Float16* P = &Pw[wid][0];
        for (int ks = 0; ks < 8; ++ks) {
            #pragma unroll
            for (int t = 0; t < 2; ++t) {
                int nt = ks * 2 + t;
                #pragma unroll
                for (int r = 0; r < 8; ++r)
                    P[(r + hl * 8) * 32 + t * 16 + l15] =
                        (_Float16)(s[nt][r] * inv[r]);
            }
            const _Float16* pr = P + l15 * 32;
            v16h ap = frag_from(pr + hl * 8, pr + 16 + hl * 8);
            #pragma unroll
            for (int j = 0; j < 2; ++j) {
                const _Float16* vr = &Vs[(j * 16 + l15) * LL + ks * 32];
                v16h bv = frag_from(vr + hl * 8, vr + 16 + hl * 8);
                o[j] = wmma16(ap, bv, o[j]);
            }
        }
        // y[b, row, h*32 + dd]  (f16, feeds the proj GEMM)
        #pragma unroll
        for (int j = 0; j < 2; ++j)
            #pragma unroll
            for (int r = 0; r < 8; ++r) {
                int row = rbase + r + hl * 8;
                y[((size_t)(b * LL + row)) * CC + h * DD + j * 16 + l15] =
                    (_Float16)o[j][r];
            }
    }
}

// ---------------------------------------------------------------------------
// Host-side orchestration
// ---------------------------------------------------------------------------
extern "C" void kernel_launch(void* const* d_in, const int* in_sizes, int n_in,
                              void* d_out, int out_size, void* d_ws, size_t ws_size,
                              hipStream_t stream) {
    (void)in_sizes; (void)n_in; (void)out_size; (void)ws_size;
    const float* x        = (const float*)d_in[0];
    const float* qkv_w    = (const float*)d_in[1];
    const float* qkv_b    = (const float*)d_in[2];
    const float* proj_w   = (const float*)d_in[3];
    const float* proj_b   = (const float*)d_in[4];
    const float* fc1_w    = (const float*)d_in[5];
    const float* fc1_b    = (const float*)d_in[6];
    const float* fc2_w    = (const float*)d_in[7];
    const float* fc2_b    = (const float*)d_in[8];
    const float* lgs      = (const float*)d_in[9];
    float* out            = (float*)d_out;

    // workspace carve-up (~271 MB; yh aliases xh after the QKV GEMM)
    char* ws = (char*)d_ws;
    size_t off = 0;
    auto alloc = [&](size_t bytes) { size_t p = off; off = (off + bytes + 255) & ~(size_t)255; return p; };
    const size_t nX   = (size_t)BW * LL * CC;            // 33.5M
    const size_t nQKVW = (size_t)NQKV * CC;
    const size_t nPW  = (size_t)CC * CC;
    const size_t nHead = (size_t)BW * HH * LL * DD;      // 33.5M per tensor
    _Float16* xh    = (_Float16*)(ws + alloc(nX * 2));
    _Float16* wqkv  = (_Float16*)(ws + alloc(nQKVW * 2));
    _Float16* wproj = (_Float16*)(ws + alloc(nPW * 2));
    _Float16* qh    = (_Float16*)(ws + alloc(nHead * 2));
    _Float16* kh    = (_Float16*)(ws + alloc(nHead * 2));
    _Float16* vT    = (_Float16*)(ws + alloc(nHead * 2));
    float*    btab  = (float*)(ws + alloc((size_t)HH * 961 * 4));
    _Float16* yh    = xh;   // x no longer needed after QKV GEMM

    // 1) converts
    convert_f16_kernel<<<(unsigned)(nX / 4 / 256), 256, 0, stream>>>(x, xh, nX / 4);
    convert_f16_kernel<<<(unsigned)(nQKVW / 4 / 256), 256, 0, stream>>>(qkv_w, wqkv, nQKVW / 4);
    convert_f16_kernel<<<(unsigned)(nPW / 4 / 256), 256, 0, stream>>>(proj_w, wproj, nPW / 4);
    // 2) bias table (961 unique (dy,dx) entries)
    bias_table_kernel<<<961, 512, 0, stream>>>(fc1_w, fc1_b, fc2_w, fc2_b, btab);
    // 3) QKV GEMM, scatter into packed q/k/vT
    gemm_f16_kernel<<<dim3((BW * LL) / 128, NQKV / 128), 256, 0, stream>>>(
        xh, wqkv, qkv_b, nullptr, qh, kh, vT, BW * LL, NQKV, CC, 1);
    // 4) cosine-attention normalization (+logit scale folded into q)
    const unsigned nrmBlocks = (unsigned)((size_t)BW * HH * LL / 8);
    normalize_kernel<<<nrmBlocks, 256, 0, stream>>>(qh, lgs, 1);
    normalize_kernel<<<nrmBlocks, 256, 0, stream>>>(kh, lgs, 0);
    // 5) fused attention (one block per (b,h))
    attn_kernel<<<BW * HH, 256, 0, stream>>>(qh, kh, vT, btab, yh);
    // 6) output projection -> f32 out
    gemm_f16_kernel<<<dim3((BW * LL) / 128, CC / 128), 256, 0, stream>>>(
        yh, wproj, proj_b, out, nullptr, nullptr, nullptr, BW * LL, CC, CC, 0);
}
